// ScaledDotProductAttention_50672024158499
// MI455X (gfx1250) — matrix operand order repaired and checked by reading, not compiled, not run on hardware
//
#include <hip/hip_runtime.h>
#include <hip/hip_bf16.h>
#include <math.h>

// ---------------------------------------------------------------------------
// Fused causal attention for gfx1250 (MI455X), wave32.
//   B=8, T=2048, E=1024, HD=64
// CDNA5 paths used:
//   - v_wmma_f32_16x16x32_f16 for all matmuls (f32 accumulate)
//   - global_load_async_to_lds_b128 + s_wait_asynccnt (ASYNCcnt) to stage
//     tiles in LDS without tying up VGPRs -> true load/WMMA overlap
//   - ds_load_b128 fragment reads, per-wave LDS transpose for layout changes
// ---------------------------------------------------------------------------

typedef _Float16 f16;
typedef __attribute__((ext_vector_type(16))) _Float16 v16h;
typedef __attribute__((ext_vector_type(8)))  _Float16 v8h;
typedef __attribute__((ext_vector_type(8)))  float    v8f;
typedef __attribute__((ext_vector_type(4)))  float    v4f;

#define B_  8
#define T_  2048
#define E_  1024
#define HD_ 64

__device__ __forceinline__ v16h hcat(v8h a, v8h b) {
  return __builtin_shufflevector(a, b, 0,1,2,3,4,5,6,7,8,9,10,11,12,13,14,15);
}
__device__ __forceinline__ v8f wmma_f16(v16h a, v16h b, v8f c) {
  return __builtin_amdgcn_wmma_f32_16x16x32_f16(false, a, false, b,
                                                (short)0, c, false, false);
}
// 16B global -> LDS async copy (ASYNCcnt), no dest VGPRs.
__device__ __forceinline__ void async_ld_b128(const f16* g, const f16* l) {
  unsigned       lo = (unsigned)(uintptr_t)l;              // LDS byte offset
  unsigned long long ga = (unsigned long long)(uintptr_t)g;
  asm volatile("global_load_async_to_lds_b128 %0, %1, off"
               :: "v"(lo), "v"(ga) : "memory");
}
#define WAIT_ASYNC(n) asm volatile("s_wait_asynccnt " #n ::: "memory")

// Reductions across 16 lanes of each wave32 half (xor masks <16 stay in-half).
__device__ __forceinline__ float redmax16(float v) {
  #pragma unroll
  for (int m = 1; m < 16; m <<= 1) v = fmaxf(v, __shfl_xor(v, m, 32));
  return v;
}
__device__ __forceinline__ float redsum16(float v) {
  #pragma unroll
  for (int m = 1; m < 16; m <<= 1) v += __shfl_xor(v, m, 32);
  return v;
}

// ---------------- 1) x fp32 -> f16 ----------------
__global__ void __launch_bounds__(256) cvt_x(const float* __restrict__ x,
                                             f16* __restrict__ xh) {
  size_t i = ((size_t)blockIdx.x * 256 + threadIdx.x) * 8;
  v4f a = *(const v4f*)(x + i);
  v4f b = *(const v4f*)(x + i + 4);
  v8h o;
  #pragma unroll
  for (int e = 0; e < 4; ++e) { o[e] = (f16)a[e]; o[e + 4] = (f16)b[e]; }
  *(v8h*)(xh + i) = o;
}

// ---------------- 2) W [E,HD] fp32 -> Wt [HD,E] f16 ----------------
__global__ void __launch_bounds__(256) cvt_w(const float* __restrict__ w0,
                                             const float* __restrict__ w1,
                                             const float* __restrict__ w2,
                                             f16* __restrict__ t0,
                                             f16* __restrict__ t1,
                                             f16* __restrict__ t2) {
  int i = blockIdx.x * 256 + threadIdx.x;     // 65536 per matrix
  int h = i >> 10, e = i & 1023;              // out index = h*E + e
  const float* w = (blockIdx.y == 0) ? w0 : (blockIdx.y == 1) ? w1 : w2;
  f16*         t = (blockIdx.y == 0) ? t0 : (blockIdx.y == 1) ? t1 : t2;
  t[i] = (f16)w[(size_t)e * HD_ + h];
}

// ---------------- 3) QKV projection, async-LDS double-buffered -------------
// block = 8 waves, all same matrix; wave = one 16-row tile.  3 x 1024 waves.
__global__ void __launch_bounds__(256) qkv_gemm(const f16* __restrict__ xh,
                                                const f16* __restrict__ wq,
                                                const f16* __restrict__ wk,
                                                const f16* __restrict__ wv,
                                                f16* __restrict__ qh,
                                                f16* __restrict__ kh,
                                                f16* __restrict__ vt) {
  __shared__ __align__(32) f16 bbuf[2][64 * 64];      // W chunk, block-shared
  __shared__ __align__(32) f16 abuf[8][2][16 * 64];   // x tile, per wave
  __shared__ __align__(32) f16 tbuf[8][16 * 64];      // store transpose tile
  const int tid  = threadIdx.x;
  const int lane = tid & 31;
  const int wid  = tid >> 5;
  const int gw   = blockIdx.x * 8 + wid;              // 3072 waves
  const int mat  = gw >> 10;                          // 0=Q 1=K 2=V (uniform per block)
  const int tile = gw & 1023;
  const int n  = lane & 15;
  const int hi = lane >> 4;
  const int g0 = tile * 16;

  const f16* w  = (mat == 0) ? wq : (mat == 1) ? wk : wv;
  const f16* xg = xh + (size_t)g0 * E_;

  // 64x64 W chunk: 512 x 16B, one per thread per t
  auto copyB = [&](int p, int k0) {
    #pragma unroll
    for (int t = 0; t < 2; ++t) {
      int s = t * 256 + tid;                 // 0..511
      int row = s >> 3, seg = s & 7;
      async_ld_b128(w + (size_t)row * E_ + k0 + seg * 8,
                    &bbuf[p][row * 64 + seg * 8]);
    }
  };
  // 16x64 x tile: 128 x 16B per wave
  auto copyA = [&](int p, int k0) {
    #pragma unroll
    for (int t = 0; t < 4; ++t) {
      int s = t * 32 + lane;                 // 0..127
      int row = s >> 3, seg = s & 7;
      async_ld_b128(xg + (size_t)row * E_ + k0 + seg * 8,
                    &abuf[wid][p][row * 64 + seg * 8]);
    }
  };

  v8f acc[4] = {};
  copyB(0, 0);
  copyA(0, 0);                               // 6 asyncs/thread outstanding

  for (int c = 0; c < 16; ++c) {             // 16 chunks of 64
    const int p = c & 1;
    if (c < 15) { copyB(p ^ 1, (c + 1) * 64); copyA(p ^ 1, (c + 1) * 64);
                  WAIT_ASYNC(6); }
    else        { WAIT_ASYNC(0); }
    __syncthreads();                         // all waves' chunk-c copies landed

    const f16* ab = &abuf[wid][p][0];
    const f16* bb = &bbuf[p][0];
    #pragma unroll
    for (int sub = 0; sub < 2; ++sub) {      // two 32-wide K sub-chunks
      const f16* ar = ab + n * 64 + sub * 32 + hi * 16;
      v16h a = *(const v16h*)(ar);
      #pragma unroll
      for (int nt = 0; nt < 4; ++nt) {
        v16h bf = *(const v16h*)(bb + (nt * 16 + n) * 64 + sub * 32 + hi * 16);
        acc[nt] = wmma_f16(a, bf, acc[nt]);
      }
    }
    __syncthreads();                         // readers done before overwrite
  }

  const int b  = g0 >> 11;
  const int t0 = g0 & 2047;
  if (mat == 2) {
    // V transposed: Vt[b][col][t], 8 consecutive t -> one 16B store per nt
    #pragma unroll
    for (int nt = 0; nt < 4; ++nt) {
      int col = nt * 16 + n;
      v8h vv;
      #pragma unroll
      for (int e = 0; e < 8; ++e) vv[e] = (f16)acc[nt][e];
      *(v8h*)(vt + (size_t)(b * HD_ + col) * T_ + t0 + 8 * hi) = vv;
    }
  } else {
    // Q/K row-major: transpose C-layout via per-wave LDS tile, then b128 out
    f16* tb = &tbuf[wid][0];
    #pragma unroll
    for (int nt = 0; nt < 4; ++nt) {
      int col = nt * 16 + n;
      #pragma unroll
      for (int e = 0; e < 8; ++e)
        tb[(e + 8 * hi) * 64 + col] = (f16)acc[nt][e];
    }
    __asm__ volatile("" ::: "memory");       // DS in-order within a wave
    v16h r = *(const v16h*)(tb + n * 64 + hi * 32);
    f16* dst = (mat == 0) ? qh : kh;
    *(v16h*)(dst + (size_t)(g0 + n) * HD_ + hi * 32) = r;
  }
}

// ---------------- 4) flash attention: one wave = 16 queries ----------------
// per-wave async double-buffered K/V tiles in LDS (barrier-free).
__global__ void __launch_bounds__(256) flash_fwd(const f16* __restrict__ qh,
                                                 const f16* __restrict__ kh,
                                                 const f16* __restrict__ vt,
                                                 float* __restrict__ out) {
  __shared__ __align__(32) f16 kbuf[8][2][32 * 64];  // 64KB
  __shared__ __align__(32) f16 vbuf[8][2][64 * 32];  // 64KB
  __shared__ __align__(16) f16 pbuf[8][16 * 32];     // 8KB
  const int lane = threadIdx.x & 31;
  const int wid  = threadIdx.x >> 5;
  const int tile = blockIdx.x * 8 + wid;             // 1024 query tiles
  const int b  = tile >> 7;
  const int q0 = (tile & 127) << 4;
  const int n  = lane & 15;
  const int hi = lane >> 4;

  // Q as A-fragments (one-time direct loads)
  const f16* qrow = qh + ((size_t)(b * T_ + q0 + n)) * HD_ + hi * 16;
  v16h qlo = *(const v16h*)(qrow);
  v16h qhh = *(const v16h*)(qrow + 32);

  const f16* kb_ = kh + (size_t)b * T_ * HD_;
  const f16* vb_ = vt + (size_t)b * HD_ * T_;

  // 32 keys x 64 f16 (K rows) -> 256 x 16B, 8 asyncs per lane
  auto copyK = [&](int p, int j) {
    #pragma unroll
    for (int t = 0; t < 8; ++t) {
      int s = t * 32 + lane;
      int row = s >> 3, seg = s & 7;
      async_ld_b128(kb_ + (size_t)(j + row) * HD_ + seg * 8,
                    &kbuf[wid][p][row * 64 + seg * 8]);
    }
  };
  // 64 head-cols x 32 keys (Vt rows) -> 256 x 16B, 8 asyncs per lane
  auto copyV = [&](int p, int j) {
    #pragma unroll
    for (int t = 0; t < 8; ++t) {
      int s = t * 32 + lane;
      int row = s >> 2, seg = s & 3;
      async_ld_b128(vb_ + (size_t)row * T_ + j + seg * 8,
                    &vbuf[wid][p][row * 32 + seg * 8]);
    }
  };

  v8f o0 = {}, o1 = {}, o2 = {}, o3 = {};
  float mrow[8], lrow[8];
  #pragma unroll
  for (int e = 0; e < 8; ++e) { mrow[e] = -__builtin_inff(); lrow[e] = 0.0f; }

  const float sc  = 0.125f;                   // HD^-0.5
  const float L2E = 1.44269504088896341f;
  const int  jend = q0 + 16;                  // causal bound

  copyK(0, 0); copyV(0, 0);                   // 16 asyncs in flight

  for (int j = 0; j < jend; j += 32) {
    const int p = (j >> 5) & 1;
    if (j + 32 < jend) { copyK(p ^ 1, j + 32); copyV(p ^ 1, j + 32);
                         WAIT_ASYNC(16); }
    else               { WAIT_ASYNC(0); }

    // K fragments from LDS (in-wave: no barrier needed after asynccnt wait)
    const f16* kb = &kbuf[wid][p][0];
    v16h kc0 = *(const v16h*)(kb + n * 64 + hi * 16);
    v16h kc1 = *(const v16h*)(kb + n * 64 + hi * 16 + 32);
    v16h kc2 = *(const v16h*)(kb + (16 + n) * 64 + hi * 16);
    v16h kc3 = *(const v16h*)(kb + (16 + n) * 64 + hi * 16 + 32);

    v8f s0 = {}, s1 = {};
    s0 = wmma_f16(qlo, kc0, s0); s0 = wmma_f16(qhh, kc1, s0);
    s1 = wmma_f16(qlo, kc2, s1); s1 = wmma_f16(qhh, kc3, s1);

    // online softmax (C layout: row = e + 8*hi, col = key = j(+16) + n)
    float p0[8], p1[8], alpha[8];
    #pragma unroll
    for (int e = 0; e < 8; ++e) {
      int row = q0 + e + 8 * hi;
      float a0 = (j + n      <= row) ? (float)s0[e] * sc : -__builtin_inff();
      float a1 = (j + 16 + n <= row) ? (float)s1[e] * sc : -__builtin_inff();
      float t    = redmax16(fmaxf(a0, a1));
      float mnew = fmaxf(mrow[e], t);
      alpha[e] = exp2f((mrow[e] - mnew) * L2E);
      p0[e]    = exp2f((a0 - mnew) * L2E);
      p1[e]    = exp2f((a1 - mnew) * L2E);
      lrow[e]  = lrow[e] * alpha[e] + redsum16(p0[e] + p1[e]);
      mrow[e]  = mnew;
    }
    #pragma unroll
    for (int e = 0; e < 8; ++e) {
      o0[e] *= alpha[e]; o1[e] *= alpha[e];
      o2[e] *= alpha[e]; o3[e] *= alpha[e];
    }

    // P : C-layout -> A-layout via per-wave LDS tile (DS in-order per wave)
    f16* pb = &pbuf[wid][0];
    #pragma unroll
    for (int e = 0; e < 8; ++e) {
      int m = e + 8 * hi;
      pb[m * 32 + n]      = (f16)p0[e];
      pb[m * 32 + n + 16] = (f16)p1[e];
    }
    __asm__ volatile("" ::: "memory");
    const f16* pr = pb + n * 32 + hi * 16;
    v16h pA = hcat(*(const v8h*)(pr), *(const v8h*)(pr + 8));
    __asm__ volatile("" ::: "memory");

    // O += P(16x32) * V(32x64), V fragments from LDS
    const f16* vb = &vbuf[wid][p][0];
    o0 = wmma_f16(pA, *(const v16h*)(vb + (0 * 16 + n) * 32 + hi * 16), o0);
    o1 = wmma_f16(pA, *(const v16h*)(vb + (1 * 16 + n) * 32 + hi * 16), o1);
    o2 = wmma_f16(pA, *(const v16h*)(vb + (2 * 16 + n) * 32 + hi * 16), o2);
    o3 = wmma_f16(pA, *(const v16h*)(vb + (3 * 16 + n) * 32 + hi * 16), o3);
  }

  // epilogue: divide by l, store fp32
  float* op = out + ((size_t)(b * T_ + q0)) * HD_;
  #pragma unroll
  for (int e = 0; e < 8; ++e) {
    float inv = 1.0f / lrow[e];
    int m = e + 8 * hi;
    op[(size_t)m * HD_ + n]      = o0[e] * inv;
    op[(size_t)m * HD_ + n + 16] = o1[e] * inv;
    op[(size_t)m * HD_ + n + 32] = o2[e] * inv;
    op[(size_t)m * HD_ + n + 48] = o3[e] * inv;
  }
}

// ---------------------------------------------------------------------------
extern "C" void kernel_launch(void* const* d_in, const int* in_sizes, int n_in,
                              void* d_out, int out_size, void* d_ws,
                              size_t ws_size, hipStream_t stream) {
  (void)in_sizes; (void)n_in; (void)out_size; (void)ws_size;
  const float* x  = (const float*)d_in[0];
  const float* Wq = (const float*)d_in[1];
  const float* Wk = (const float*)d_in[2];
  const float* Wv = (const float*)d_in[3];
  float* out = (float*)d_out;

  char* ws = (char*)d_ws;
  size_t off = 0;
  f16* xh  = (f16*)(ws + off); off += (size_t)B_ * T_ * E_ * sizeof(f16);
  f16* wtq = (f16*)(ws + off); off += (size_t)HD_ * E_ * sizeof(f16);
  f16* wtk = (f16*)(ws + off); off += (size_t)HD_ * E_ * sizeof(f16);
  f16* wtv = (f16*)(ws + off); off += (size_t)HD_ * E_ * sizeof(f16);
  f16* qh  = (f16*)(ws + off); off += (size_t)B_ * T_ * HD_ * sizeof(f16);
  f16* kh  = (f16*)(ws + off); off += (size_t)B_ * T_ * HD_ * sizeof(f16);
  f16* vt  = (f16*)(ws + off); off += (size_t)B_ * T_ * HD_ * sizeof(f16);

  cvt_x<<<dim3(8192), 256, 0, stream>>>(x, xh);
  cvt_w<<<dim3(256, 3), 256, 0, stream>>>(Wq, Wk, Wv, wtq, wtk, wtv);
  qkv_gemm<<<dim3(384), 256, 0, stream>>>(xh, wtq, wtk, wtv, qh, kh, vt);
  flash_fwd<<<dim3(128), 256, 0, stream>>>(qh, kh, vt, out);
}
